// AttnBlockpp_54674933678741
// MI455X (gfx1250) — compile-verified, matrix operand order checked by reading, not run on hardware
//
#include <hip/hip_runtime.h>
#include <hip/hip_bf16.h>
#include <stdint.h>

typedef __attribute__((ext_vector_type(16))) _Float16 v16h;
typedef __attribute__((ext_vector_type(8)))  _Float16 v8h;
typedef __attribute__((ext_vector_type(8)))  float    v8f;

#define B_  8
#define C_  256
#define L_  2048
#define G_  32
#define NPG 16384   // (C_/G_)*L_ elements per (b,g) group

// ---- WMMA fragment loader -------------------------------------------------
// 16-bit A/B fragment for 16x16x32: row picked by caller (M or N = lane&15),
// K mapping per ISA: lanes 0-15 hold K = {0..7, 16..23}, lanes 16-31 hold
// K = {8..15, 24..31}  -> two contiguous 16-byte chunks per lane.
static __device__ __forceinline__ v16h ldfrag(const _Float16* rowptr, int c0, int lane) {
  const int hi = (lane & 16) ? 8 : 0;
  union { v16h v; v8h h[2]; } u;
  u.h[0] = *(const v8h*)(rowptr + c0 + hi);
  u.h[1] = *(const v8h*)(rowptr + c0 + hi + 16);
  return u.v;
}

static __device__ __forceinline__ v8f wmma_f16(v16h a, v16h b, v8f c) {
  return __builtin_amdgcn_wmma_f32_16x16x32_f16(false, a, false, b, (short)0, c, false, false);
}

// ---- CDNA5 async global->LDS copy ----------------------------------------
static __device__ __forceinline__ uint32_t lds_addr_of(const void* p) {
  return (uint32_t)(uintptr_t)p;   // low 32 bits of generic LDS pointer = LDS byte address
}
static __device__ __forceinline__ void async_cp_b128(uint32_t lds_addr, const void* gptr) {
  asm volatile("global_load_async_to_lds_b128 %0, %1, off"
               :: "v"(lds_addr), "v"(gptr) : "memory");
}
static __device__ __forceinline__ void wait_async0() {
  asm volatile("s_wait_asynccnt 0x0" ::: "memory");
}
static __device__ __forceinline__ void wait_async16() {
  asm volatile("s_wait_asynccnt 0x10" ::: "memory");
}

// ---- 0: weights fp32 (C,D) -> f16 transposed (D,C) ------------------------
__global__ void wcvt_kernel(const float* __restrict__ w0, const float* __restrict__ w1,
                            const float* __restrict__ w2, const float* __restrict__ w3,
                            _Float16* __restrict__ wt) {
  const int e = blockIdx.x * 256 + threadIdx.x;   // 0 .. 4*65536-1
  const int sel = e >> 16;
  const int r = e & 65535;
  const int d = r >> 8, c = r & 255;
  const float* w = sel == 0 ? w0 : sel == 1 ? w1 : sel == 2 ? w2 : w3;
  wt[(size_t)sel * 65536 + d * 256 + c] = (_Float16)w[c * 256 + d];
}

// ---- 1: GroupNorm statistics ----------------------------------------------
__global__ void gn_stats_kernel(const float* __restrict__ x,
                                float* __restrict__ mean, float* __restrict__ rstd) {
  const int bg = blockIdx.x;               // B_*G_ = 256 blocks
  const int b = bg >> 5, g = bg & 31;
  const float* xp = x + ((size_t)b * C_ + g * 8) * L_;  // 8 contiguous channels
  float s = 0.f, sq = 0.f;
  for (int i = threadIdx.x; i < NPG; i += 256) {
    float v = xp[i];
    s += v; sq += v * v;
  }
  __shared__ float ss[256], sv[256];
  ss[threadIdx.x] = s; sv[threadIdx.x] = sq;
  __syncthreads();
  for (int st = 128; st > 0; st >>= 1) {
    if (threadIdx.x < st) { ss[threadIdx.x] += ss[threadIdx.x + st];
                            sv[threadIdx.x] += sv[threadIdx.x + st]; }
    __syncthreads();
  }
  if (threadIdx.x == 0) {
    float m = ss[0] * (1.f / NPG);
    float var = sv[0] * (1.f / NPG) - m * m;
    mean[bg] = m;
    rstd[bg] = rsqrtf(var + 1e-6f);
  }
}

// ---- 2: apply GN, write h16 in (B,L,C) f16 --------------------------------
__global__ void gn_apply_kernel(const float* __restrict__ x,
                                const float* __restrict__ gamma, const float* __restrict__ beta,
                                const float* __restrict__ mean, const float* __restrict__ rstd,
                                _Float16* __restrict__ h16) {
  const int e = (blockIdx.x * 256 + threadIdx.x) * 4;
  const int b = e / (C_ * L_);
  const int rem = e - b * (C_ * L_);
  const int c = rem >> 11;
  const int l = rem & (L_ - 1);
  const int bg = b * G_ + (c >> 3);
  const float r = rstd[bg];
  const float ga = gamma[c] * r;
  const float be = beta[c] - mean[bg] * ga;
  const float4 xv = *(const float4*)(x + e);
  _Float16* hp = h16 + ((size_t)b * L_ + l) * C_ + c;   // transpose store
  hp[0]      = (_Float16)(xv.x * ga + be);
  hp[C_]     = (_Float16)(xv.y * ga + be);
  hp[2 * C_] = (_Float16)(xv.z * ga + be);
  hp[3 * C_] = (_Float16)(xv.w * ga + be);
}

// ---- 3: fused Q/K/V projections: weights staged in LDS via async DMA ------
// q[d,l] = sum_c w0t[d,c] h[l,c] (+bias) ; Q scaled by C^-0.5 = 1/16.
// Q,K stored (B,L,C) f16; V stored (B,C,L) f16.
__global__ void __launch_bounds__(128) proj_qkv_kernel(
    const _Float16* __restrict__ h16, const _Float16* __restrict__ wt,
    const float* __restrict__ b0, const float* __restrict__ b1, const float* __restrict__ b2,
    _Float16* __restrict__ q16, _Float16* __restrict__ k16, _Float16* __restrict__ v16) {
  __shared__ _Float16 Wl[65536];          // 128 KB: current weight matrix (D,C) f16

  const int tid  = threadIdx.x;
  const int lane = tid & 31;
  const int wave = tid >> 5;
  const int b  = blockIdx.x >> 5;
  const int l0 = (blockIdx.x & 31) * 64 + wave * 16;
  const int ln = lane & 15;
  const int hi8 = (lane & 16) ? 8 : 0;

  const _Float16* hrow = h16 + ((size_t)b * L_ + l0 + ln) * C_;
  v16h hfr[8];
#pragma unroll
  for (int kk = 0; kk < 8; ++kk) hfr[kk] = ldfrag(hrow, kk * 32, lane);

  for (int mt = 0; mt < 3; ++mt) {
    const _Float16* w = wt + (size_t)mt * 65536;
    const float* bias = mt == 0 ? b0 : mt == 1 ? b1 : b2;
    const float scale = (mt == 0) ? 0.0625f : 1.0f;

    // stage whole 256x256 f16 weight matrix into LDS (8192 b128, 64/thread)
    for (int j = 0; j < 64; ++j) {
      const int ck = tid + j * 128;
      async_cp_b128(lds_addr_of(&Wl[ck * 8]), w + ck * 8);
    }
    wait_async0();
    __syncthreads();

    for (int dt = 0; dt < 16; dt += 2) {
      const int d0 = dt * 16;
      const int d1 = d0 + 16;
      v8f acc0 = {}, acc1 = {};
#pragma unroll
      for (int kk = 0; kk < 8; ++kk) {
        v16h a0 = ldfrag(&Wl[(d0 + ln) * 256], kk * 32, lane);
        v16h a1 = ldfrag(&Wl[(d1 + ln) * 256], kk * 32, lane);
        acc0 = wmma_f16(a0, hfr[kk], acc0);
        acc1 = wmma_f16(a1, hfr[kk], acc1);
      }
      if (mt < 2) {
        _Float16* outp = (mt == 0 ? q16 : k16);
        _Float16* rowp = outp + ((size_t)b * L_ + l0 + ln) * C_;
        v8h buf0, buf1;
#pragma unroll
        for (int r = 0; r < 8; ++r) {
          buf0[r] = (_Float16)((acc0[r] + bias[d0 + hi8 + r]) * scale);
          buf1[r] = (_Float16)((acc1[r] + bias[d1 + hi8 + r]) * scale);
        }
        *(v8h*)(rowp + d0 + hi8) = buf0;
        *(v8h*)(rowp + d1 + hi8) = buf1;
      } else {
#pragma unroll
        for (int r = 0; r < 8; ++r) {
          const int da = d0 + hi8 + r;
          const int db = d1 + hi8 + r;
          v16[((size_t)b * C_ + da) * L_ + l0 + ln] = (_Float16)(acc0[r] + bias[da]);
          v16[((size_t)b * C_ + db) * L_ + l0 + ln] = (_Float16)(acc1[r] + bias[db]);
        }
      }
    }
    __syncthreads();   // all waves done with Wl before next matrix overwrites
  }
}

// ---- 4: flash attention, double-buffered async LDS staging + WMMA ---------
__global__ void __launch_bounds__(128) attn_kernel(
    const _Float16* __restrict__ q16, const _Float16* __restrict__ k16,
    const _Float16* __restrict__ v16, _Float16* __restrict__ a16) {
  __shared__ _Float16 Kt[2][32 * 256];   // [buf][key i][c]   2 x 16 KB
  __shared__ _Float16 Vt[2][256 * 32];   // [buf][c][key i]   2 x 16 KB
  __shared__ _Float16 Pt[4][16 * 32];    // per-wave probs [l][i]
  __shared__ float    stats[4][16];      // per-wave row stats

  const int tid  = threadIdx.x;
  const int lane = tid & 31;
  const int wave = tid >> 5;
  const int b  = blockIdx.x >> 5;
  const int l0 = (blockIdx.x & 31) * 64 + wave * 16;
  const int ln = lane & 15;

  const _Float16* qrow = q16 + ((size_t)b * L_ + l0 + ln) * C_;
  v16h qf[8];
#pragma unroll
  for (int kk = 0; kk < 8; ++kk) qf[kk] = ldfrag(qrow, kk * 32, lane);

  v8f O[16];
#pragma unroll
  for (int t = 0; t < 16; ++t) O[t] = {};
  float mrow[8], lrow[8];
#pragma unroll
  for (int r = 0; r < 8; ++r) { mrow[r] = -3.0e38f; lrow[r] = 0.f; }

  // issue a K/V tile (16 async b128 per thread) into buffer `buf`
  auto issue_tile = [&](int i0, int buf) {
#pragma unroll
    for (int j = 0; j < 8; ++j) {
      int ck = tid + j * 128;
      int kr = ck >> 5, kc = (ck & 31) * 8;
      async_cp_b128(lds_addr_of(&Kt[buf][kr * 256 + kc]),
                    k16 + ((size_t)b * L_ + i0 + kr) * C_ + kc);
      int vr = ck >> 2, vc = (ck & 3) * 8;
      async_cp_b128(lds_addr_of(&Vt[buf][vr * 32 + vc]),
                    v16 + ((size_t)b * C_ + vr) * L_ + i0 + vc);
    }
  };

  const int NB = L_ / 32;
  issue_tile(0, 0);

  for (int ib = 0; ib < NB; ++ib) {
    const int cur = ib & 1;
    // prefetch next tile into the other buffer, then wait for current tile
    if (ib + 1 < NB) {
      issue_tile((ib + 1) * 32, cur ^ 1);
      wait_async16();        // first 16 (current tile) complete; next 16 in flight
    } else {
      wait_async0();
    }
    __syncthreads();

    const _Float16* Kb = &Kt[cur][0];
    const _Float16* Vb = &Vt[cur][0];

    // S = Q^T K  (scale already folded into Q); two independent chains
    v8f s0 = {}, s1 = {};
#pragma unroll
    for (int kk = 0; kk < 8; ++kk) {
      v16h bk0 = ldfrag(Kb + ln * 256, kk * 32, lane);
      v16h bk1 = ldfrag(Kb + (ln + 16) * 256, kk * 32, lane);
      s0 = wmma_f16(qf[kk], bk0, s0);
      s1 = wmma_f16(qf[kk], bk1, s1);
    }

    // online softmax; row l = r (lanes 0-15) / r+8 (lanes 16-31)
#pragma unroll
    for (int r = 0; r < 8; ++r) {
      float mx = fmaxf(s0[r], s1[r]);
      mx = fmaxf(mx, __shfl_xor(mx, 1, 16));
      mx = fmaxf(mx, __shfl_xor(mx, 2, 16));
      mx = fmaxf(mx, __shfl_xor(mx, 4, 16));
      mx = fmaxf(mx, __shfl_xor(mx, 8, 16));
      float mnew = fmaxf(mrow[r], mx);
      float corr = __expf(mrow[r] - mnew);
      float p0 = __expf(s0[r] - mnew);
      float p1 = __expf(s1[r] - mnew);
      float rs = p0 + p1;
      rs += __shfl_xor(rs, 1, 16);
      rs += __shfl_xor(rs, 2, 16);
      rs += __shfl_xor(rs, 4, 16);
      rs += __shfl_xor(rs, 8, 16);
      lrow[r] = lrow[r] * corr + rs;
      mrow[r] = mnew;
      const int li = (lane & 16) ? (r + 8) : r;
      if (ln == r) stats[wave][li] = corr;
      Pt[wave][li * 32 + ln]      = (_Float16)p0;
      Pt[wave][li * 32 + ln + 16] = (_Float16)p1;
    }

    // rescale O columns by correction of their query row (col N = ln)
    const float cf = stats[wave][ln];
#pragma unroll
    for (int t = 0; t < 16; ++t)
#pragma unroll
      for (int r = 0; r < 8; ++r) O[t][r] *= cf;

    // O += V * P^T   (A = V[c,i] 16x32, B = P^T[i,l] 32x16)
    v16h bp = ldfrag(&Pt[wave][ln * 32], 0, lane);
#pragma unroll
    for (int t = 0; t < 16; ++t) {
      v16h av = ldfrag(Vb + (t * 16 + ln) * 32, 0, lane);
      O[t] = wmma_f16(av, bp, O[t]);
    }
    __syncthreads();   // buffer `cur` free for tile ib+2
  }

  // normalize by row sums and store (B,L,C) f16
#pragma unroll
  for (int r = 0; r < 8; ++r) {
    const int li = (lane & 16) ? (r + 8) : r;
    if (ln == r) stats[wave][li] = lrow[r];
  }
  const float inv = 1.f / stats[wave][ln];
  const int hi8 = (lane & 16) ? 8 : 0;
#pragma unroll
  for (int t = 0; t < 16; ++t) {
    v8h buf;
#pragma unroll
    for (int r = 0; r < 8; ++r) buf[r] = (_Float16)(O[t][r] * inv);
    *(v8h*)(a16 + ((size_t)b * L_ + l0 + ln) * C_ + t * 16 + hi8) = buf;
  }
}

// ---- 5: output projection + bias + residual (fp32 out, (B,C,L)) ----------
__global__ void __launch_bounds__(128) outproj_kernel(
    const _Float16* __restrict__ a16, const _Float16* __restrict__ w3t,
    const float* __restrict__ b3, const float* __restrict__ x, float* __restrict__ out) {
  __shared__ _Float16 Wl[65536];          // 128 KB: w3t staged once

  const int tid  = threadIdx.x;
  const int lane = tid & 31;
  const int wave = tid >> 5;
  const int b  = blockIdx.x >> 5;
  const int l0 = (blockIdx.x & 31) * 64 + wave * 16;
  const int ln = lane & 15;
  const int hi8 = (lane & 16) ? 8 : 0;

  // stage weight matrix into LDS with the async DMA engine
  for (int j = 0; j < 64; ++j) {
    const int ck = tid + j * 128;
    async_cp_b128(lds_addr_of(&Wl[ck * 8]), w3t + ck * 8);
  }

  // overlap: load attention-output fragments from global while DMA runs
  const _Float16* arow = a16 + ((size_t)b * L_ + l0 + ln) * C_;
  v16h af[8];
#pragma unroll
  for (int kk = 0; kk < 8; ++kk) af[kk] = ldfrag(arow, kk * 32, lane);

  wait_async0();
  __syncthreads();

  for (int dt = 0; dt < 16; dt += 2) {
    const int d0 = dt * 16;
    const int d1 = d0 + 16;
    // start pulling residual x rows for this dt-pair toward the WGP
    __builtin_prefetch(x + ((size_t)b * C_ + d0 + hi8) * L_ + l0 + ln, 0, 1);

    v8f acc0 = {}, acc1 = {};
#pragma unroll
    for (int kk = 0; kk < 8; ++kk) {
      v16h a0 = ldfrag(&Wl[(d0 + ln) * 256], kk * 32, lane);
      v16h a1 = ldfrag(&Wl[(d1 + ln) * 256], kk * 32, lane);
      acc0 = wmma_f16(a0, af[kk], acc0);
      acc1 = wmma_f16(a1, af[kk], acc1);
    }
#pragma unroll
    for (int r = 0; r < 8; ++r) {
      const int da = d0 + hi8 + r;
      const int db = d1 + hi8 + r;
      const size_t ia = ((size_t)b * C_ + da) * L_ + l0 + ln;
      const size_t ib2 = ((size_t)b * C_ + db) * L_ + l0 + ln;
      out[ia]  = x[ia]  + acc0[r] + b3[da];
      out[ib2] = x[ib2] + acc1[r] + b3[db];
    }
  }
}

// ---- launch ---------------------------------------------------------------
extern "C" void kernel_launch(void* const* d_in, const int* in_sizes, int n_in,
                              void* d_out, int out_size, void* d_ws, size_t ws_size,
                              hipStream_t stream) {
  (void)in_sizes; (void)n_in; (void)out_size; (void)ws_size;
  const float* x     = (const float*)d_in[0];
  const float* gamma = (const float*)d_in[1];
  const float* beta  = (const float*)d_in[2];
  const float* w0    = (const float*)d_in[3];
  const float* b0    = (const float*)d_in[4];
  const float* w1    = (const float*)d_in[5];
  const float* b1    = (const float*)d_in[6];
  const float* w2    = (const float*)d_in[7];
  const float* b2    = (const float*)d_in[8];
  const float* w3    = (const float*)d_in[9];
  const float* b3    = (const float*)d_in[10];
  float* out = (float*)d_out;

  char* ws = (char*)d_ws;
  float*    mean = (float*)ws;                                  // 256 f32
  float*    rstd = (float*)(ws + 1024);                         // 256 f32
  _Float16* wt   = (_Float16*)(ws + 2048);                      // 4 x 64K f16
  _Float16* h16  = (_Float16*)(ws + 2048 + 4 * 65536 * 2);      // (B,L,C) f16, reused as a16
  _Float16* q16  = h16 + (size_t)B_ * L_ * C_;                  // (B,L,C) f16
  _Float16* k16  = q16 + (size_t)B_ * L_ * C_;                  // (B,L,C) f16
  _Float16* v16  = k16 + (size_t)B_ * L_ * C_;                  // (B,C,L) f16
  _Float16* a16  = h16;                                         // alias: h16 dead after proj

  wcvt_kernel    <<<1024, 256, 0, stream>>>(w0, w1, w2, w3, wt);
  gn_stats_kernel<<<B_ * G_, 256, 0, stream>>>(x, mean, rstd);
  gn_apply_kernel<<<(B_ * C_ * L_) / 1024, 256, 0, stream>>>(x, gamma, beta, mean, rstd, h16);
  proj_qkv_kernel<<<B_ * (L_ / 64), 128, 0, stream>>>(h16, wt, b0, b1, b2, q16, k16, v16);
  attn_kernel    <<<B_ * (L_ / 64), 128, 0, stream>>>(q16, k16, v16, a16);
  outproj_kernel <<<B_ * (L_ / 64), 128, 0, stream>>>(a16, wt + 3 * 65536, b3, x, out);
}
